// Qwen3MoeWrappedSparseMoeBlock_76287209111679
// MI455X (gfx1250) — compile-verified, matrix operand order checked by reading, not run on hardware
//
#include <hip/hip_runtime.h>
#include <hip/hip_bf16.h>

#define Hdim 2048
#define Fdim 768
#define Edim 32
#define Tn   4096
#define OUT_MAIN (Tn * Hdim)        // 8,388,608 floats
#define MT64 (Tn / 64)              // 64 token-tiles per expert (max)
#define BSTRIDE 40                  // LDS row stride (elements): 80 B, 16B-aligned, conflict-free

typedef __attribute__((ext_vector_type(16))) __bf16 v16bf;
typedef __attribute__((ext_vector_type(8)))  float  v8f;

union Frag {
    v16bf v;
    unsigned short u[16];
    uint4 q[2];
};

__device__ __forceinline__ unsigned short f2bf(float f) {
    unsigned u = __float_as_uint(f);
    u += 0x7FFFu + ((u >> 16) & 1u);   // round-to-nearest-even
    return (unsigned short)(u >> 16);
}

// ---------------------------------------------------------------- zero
__global__ void moe_zero(float* __restrict__ out, int* __restrict__ cnt, int n) {
    int i = blockIdx.x * blockDim.x + threadIdx.x;
    if (i < n) out[i] = 0.0f;
    if (i < Edim) cnt[i] = 0;
}

// ---------------------------------------------------------------- x -> bf16
__global__ void moe_cvt(const float* __restrict__ x, unsigned short* __restrict__ xb, int n) {
    int i = blockIdx.x * blockDim.x + threadIdx.x;
    if (i < n) xb[i] = f2bf(x[i]);
}

// ---------------------------------------------------------------- router (1 wave32 per token, lane == expert)
__global__ __launch_bounds__(32) void moe_router(
    const float* __restrict__ x, const float* __restrict__ gw,
    float* __restrict__ logits, float* __restrict__ topw,
    int* __restrict__ elist, int* __restrict__ cnt)
{
    int t = blockIdx.x;
    int e = threadIdx.x;               // 0..31

    const float4* xr = (const float4*)(x  + (size_t)t * Hdim);
    const float4* gr = (const float4*)(gw + (size_t)e * Hdim);
    float acc = 0.0f;
    for (int i = 0; i < Hdim / 4; ++i) {
        float4 a = xr[i], b = gr[i];
        acc += a.x * b.x + a.y * b.y + a.z * b.z + a.w * b.w;
    }
    logits[t * Edim + e] = acc;        // raw router logits -> output tuple element 1

    // softmax over the wave (E == wave32)
    float m = acc;
    for (int off = 16; off; off >>= 1) m = fmaxf(m, __shfl_xor(m, off));
    float p = __expf(acc - m);
    float s = p;
    for (int off = 16; off; off >>= 1) s += __shfl_xor(s, off);
    float prob = p / s;

    // top-4 by repeated wave argmax (tie-break: lowest index, matches lax.top_k)
    float pv = prob;
    int   selI[4]; float selW[4]; float s4 = 0.0f;
    for (int k = 0; k < 4; ++k) {
        float v = pv; int idx = e;
        for (int off = 16; off; off >>= 1) {
            float ov = __shfl_xor(v, off);
            int   oi = __shfl_xor(idx, off);
            if (ov > v || (ov == v && oi < idx)) { v = ov; idx = oi; }
        }
        selI[k] = idx; selW[k] = v; s4 += v;
        if (e == idx) pv = -1.0f;
    }
    if (e < 4) {
        float w = selW[e] * __builtin_amdgcn_rcpf(s4);   // renormalized top-k prob
        topw[t * 4 + e] = w;
        int ex = selI[e];
        int slot = atomicAdd(&cnt[ex], 1);
        elist[ex * Tn + slot] = t * 4 + e;   // packed (token, rank)
    }
}

// ---------------------------------------------------------------- phase A: hdn = silu(x@wg) * (x@wu) * route_w
// Block = 64 tokens x 16 F-cols, 4 waves. Weight K-tile (32x16) staged to LDS
// transposed (bf16, double-buffered); each thread loads ONE float4 along f
// (fully coalesced), all waves consume via ds_load_b128. One barrier / K-step.
__global__ __launch_bounds__(128) void moe_gateup(
    const float* __restrict__ wg, const float* __restrict__ wu,
    const unsigned short* __restrict__ xb,
    const int* __restrict__ elist, const int* __restrict__ cnt,
    const float* __restrict__ topw, unsigned short* __restrict__ hd)
{
    int e  = blockIdx.x / MT64;        // expert-major: L2 reuse of this expert's weights
    int mt = blockIdx.x % MT64;
    int c  = cnt[e];
    if (mt * 64 >= c) return;

    __shared__ int   rows[64];
    __shared__ float rw[64];
    __shared__ unsigned short bgT[2][16 * BSTRIDE];   // [buf][col][k], transposed
    __shared__ unsigned short buT[2][16 * BSTRIDE];

    int tid = threadIdx.x;
    if (tid < 64) {
        int slot = mt * 64 + tid;
        if (slot < c) { int v = elist[e * Tn + slot]; rows[tid] = v; rw[tid] = topw[v]; }
        else          { rows[tid] = -1; rw[tid] = 0.0f; }
    }

    int lane = tid & 31, w = tid >> 5;
    int hi   = lane >> 4;              // half-wave
    int koff = hi * 8;
    int n    = lane & 15;
    int fcol = blockIdx.y * 16 + n;

    // staging: tid -> (k0 = tid>>2, c0 = (tid&3)*4); one float4 per matrix
    int k0 = tid >> 2;                 // 0..31
    int c0 = (tid & 3) * 4;            // 0,4,8,12
    const size_t wofs = (size_t)e * Hdim * Fdim + (size_t)k0 * Fdim
                      + (size_t)blockIdx.y * 16 + c0;

    // prime buffer 0 (kb = 0)
    {
        float4 g4 = *(const float4*)(wg + wofs);
        float4 u4 = *(const float4*)(wu + wofs);
        bgT[0][(c0 + 0) * BSTRIDE + k0] = f2bf(g4.x);
        bgT[0][(c0 + 1) * BSTRIDE + k0] = f2bf(g4.y);
        bgT[0][(c0 + 2) * BSTRIDE + k0] = f2bf(g4.z);
        bgT[0][(c0 + 3) * BSTRIDE + k0] = f2bf(g4.w);
        buT[0][(c0 + 0) * BSTRIDE + k0] = f2bf(u4.x);
        buT[0][(c0 + 1) * BSTRIDE + k0] = f2bf(u4.y);
        buT[0][(c0 + 2) * BSTRIDE + k0] = f2bf(u4.z);
        buT[0][(c0 + 3) * BSTRIDE + k0] = f2bf(u4.w);
    }

    int va = rows[w * 16 + (lane & 15)];          // rows[] written by tid<64 above;
    size_t arow = (size_t)((va < 0 ? 0 : va) >> 2) * Hdim;  // read before the first __syncthreads below

    v8f cg = {}; v8f cu = {};
    int p = 0;
    for (int kb = 0; kb < Hdim; kb += 32, p ^= 1) {
        __syncthreads();               // buffer p now visible to all waves
        if (kb + 32 < Hdim) {          // stage next tile into buffer p^1
            size_t gi = wofs + (size_t)(kb + 32) * Fdim;
            float4 g4 = *(const float4*)(wg + gi);
            float4 u4 = *(const float4*)(wu + gi);
            bgT[p ^ 1][(c0 + 0) * BSTRIDE + k0] = f2bf(g4.x);
            bgT[p ^ 1][(c0 + 1) * BSTRIDE + k0] = f2bf(g4.y);
            bgT[p ^ 1][(c0 + 2) * BSTRIDE + k0] = f2bf(g4.z);
            bgT[p ^ 1][(c0 + 3) * BSTRIDE + k0] = f2bf(g4.w);
            buT[p ^ 1][(c0 + 0) * BSTRIDE + k0] = f2bf(u4.x);
            buT[p ^ 1][(c0 + 1) * BSTRIDE + k0] = f2bf(u4.y);
            buT[p ^ 1][(c0 + 2) * BSTRIDE + k0] = f2bf(u4.z);
            buT[p ^ 1][(c0 + 3) * BSTRIDE + k0] = f2bf(u4.w);
        }
        if (kb + 64 < Hdim) {          // prefetch K-tile after next (global_prefetch_b8)
            __builtin_prefetch(wg + wofs + (size_t)(kb + 64) * Fdim, 0, 1);
            __builtin_prefetch(wu + wofs + (size_t)(kb + 64) * Fdim, 0, 1);
        }

        Frag a, bg, bu;
        const uint4* ap = (const uint4*)(xb + arow + kb + koff);
        a.q[0] = ap[0];                // K = kb+koff .. +7
        a.q[1] = ap[2];                // K = kb+koff+16 .. +23
        const uint4* bp = (const uint4*)(&bgT[p][n * BSTRIDE + koff]);
        bg.q[0] = bp[0]; bg.q[1] = bp[2];
        const uint4* up = (const uint4*)(&buT[p][n * BSTRIDE + koff]);
        bu.q[0] = up[0]; bu.q[1] = up[2];

        cg = __builtin_amdgcn_wmma_f32_16x16x32_bf16(false, a.v, false, bg.v, (short)0, cg, false, false);
        cu = __builtin_amdgcn_wmma_f32_16x16x32_bf16(false, a.v, false, bu.v, (short)0, cu, false, false);
    }

    // epilogue: hoist row metadata, then silu(g)*u*route_w
    int   vmr[8];
    float rwr[8];
    int base = w * 16 + hi * 8;
#pragma unroll
    for (int r = 0; r < 8; ++r) { vmr[r] = rows[base + r]; rwr[r] = rw[base + r]; }
#pragma unroll
    for (int r = 0; r < 8; ++r) {
        if (vmr[r] >= 0) {
            float g = cg[r], u = cu[r];
            float sig = __builtin_amdgcn_rcpf(1.0f + __expf(-g));
            float h = g * sig * u * rwr[r];
            hd[(size_t)vmr[r] * Fdim + fcol] = f2bf(h);
        }
    }
}

// ---------------------------------------------------------------- phase B: out += hdn @ wd  (weight already folded)
__global__ __launch_bounds__(128) void moe_down(
    const float* __restrict__ wd, const unsigned short* __restrict__ hd,
    const int* __restrict__ elist, const int* __restrict__ cnt,
    float* __restrict__ out)
{
    int e  = blockIdx.x / MT64;
    int mt = blockIdx.x % MT64;
    int c  = cnt[e];
    if (mt * 64 >= c) return;

    __shared__ int rows[64];
    __shared__ unsigned short bT[2][16 * BSTRIDE];

    int tid = threadIdx.x;
    if (tid < 64) {
        int slot = mt * 64 + tid;
        rows[tid] = (slot < c) ? elist[e * Tn + slot] : -1;
    }

    int lane = tid & 31, w = tid >> 5;
    int hi   = lane >> 4;
    int koff = hi * 8;
    int n    = lane & 15;
    int hcol = blockIdx.y * 16 + n;

    int k0 = tid >> 2;
    int c0 = (tid & 3) * 4;
    const size_t wofs = (size_t)e * Fdim * Hdim + (size_t)k0 * Hdim
                      + (size_t)blockIdx.y * 16 + c0;

    {
        float4 d4 = *(const float4*)(wd + wofs);
        bT[0][(c0 + 0) * BSTRIDE + k0] = f2bf(d4.x);
        bT[0][(c0 + 1) * BSTRIDE + k0] = f2bf(d4.y);
        bT[0][(c0 + 2) * BSTRIDE + k0] = f2bf(d4.z);
        bT[0][(c0 + 3) * BSTRIDE + k0] = f2bf(d4.w);
    }

    int va = rows[w * 16 + (lane & 15)];
    size_t arow = (size_t)(va < 0 ? 0 : va) * Fdim;            // hdn row (packed t*4+k)

    v8f cc = {};
    int p = 0;
    for (int kb = 0; kb < Fdim; kb += 32, p ^= 1) {
        __syncthreads();
        if (kb + 32 < Fdim) {
            float4 d4 = *(const float4*)(wd + wofs + (size_t)(kb + 32) * Hdim);
            bT[p ^ 1][(c0 + 0) * BSTRIDE + k0] = f2bf(d4.x);
            bT[p ^ 1][(c0 + 1) * BSTRIDE + k0] = f2bf(d4.y);
            bT[p ^ 1][(c0 + 2) * BSTRIDE + k0] = f2bf(d4.z);
            bT[p ^ 1][(c0 + 3) * BSTRIDE + k0] = f2bf(d4.w);
        }
        if (kb + 64 < Fdim) {
            __builtin_prefetch(wd + wofs + (size_t)(kb + 64) * Hdim, 0, 1);
        }

        Frag a, b;
        const uint4* ap = (const uint4*)(hd + arow + kb + koff);
        a.q[0] = ap[0];
        a.q[1] = ap[2];
        const uint4* bp = (const uint4*)(&bT[p][n * BSTRIDE + koff]);
        b.q[0] = bp[0]; b.q[1] = bp[2];

        cc = __builtin_amdgcn_wmma_f32_16x16x32_bf16(false, a.v, false, b.v, (short)0, cc, false, false);
    }

    int vmr[8];
    int base = w * 16 + hi * 8;
#pragma unroll
    for (int r = 0; r < 8; ++r) vmr[r] = rows[base + r];
#pragma unroll
    for (int r = 0; r < 8; ++r) {
        if (vmr[r] >= 0) {
            int t = vmr[r] >> 2;
            atomicAdd(&out[(size_t)t * Hdim + hcol], cc[r]);
        }
    }
}

// ---------------------------------------------------------------- launch
extern "C" void kernel_launch(void* const* d_in, const int* in_sizes, int n_in,
                              void* d_out, int out_size, void* d_ws, size_t ws_size,
                              hipStream_t stream) {
    const float* x  = (const float*)d_in[0];   // [B,S,H]
    const float* gw = (const float*)d_in[1];   // [E,H]
    const float* wg = (const float*)d_in[2];   // [E,H,F]
    const float* wu = (const float*)d_in[3];   // [E,H,F]
    const float* wd = (const float*)d_in[4];   // [E,F,H]

    float* out    = (float*)d_out;             // [T,H]
    float* logits = out + OUT_MAIN;            // [T,E]

    // workspace layout (256B-aligned offsets), total ~42.5 MB
    char* ws = (char*)d_ws;
    unsigned short* xb   = (unsigned short*)(ws);                        // 16,777,216 B
    float*          topw = (float*)(ws + 16777216);                      //     65,536 B
    int*            elist= (int*)  (ws + 16842752);                      //    524,288 B
    int*            cnt  = (int*)  (ws + 17367040);                      //        256 B
    unsigned short* hd   = (unsigned short*)(ws + 17367296);             // 25,165,824 B

    const int NX = OUT_MAIN;                    // 8,388,608 elements

    moe_zero<<<(NX + 255) / 256, 256, 0, stream>>>(out, cnt, NX);
    moe_cvt <<<(NX + 255) / 256, 256, 0, stream>>>(x, xb, NX);
    moe_router<<<Tn, 32, 0, stream>>>(x, gw, logits, topw, elist, cnt);
    moe_gateup<<<dim3(Edim * MT64, Fdim / 16), 128, 0, stream>>>(wg, wu, xb, elist, cnt, topw, hd);
    moe_down  <<<dim3(Edim * MT64, Hdim / 16), 128, 0, stream>>>(wd, hd, elist, cnt, out);

    (void)in_sizes; (void)n_in; (void)out_size; (void)ws_size;
}
